// CrossAttention_23347442221591
// MI455X (gfx1250) — compile-verified, matrix operand order checked by reading, not run on hardware
//
#include <hip/hip_runtime.h>

// CDNA5 / gfx1250, wave32. Matrix ops via v_wmma_f32_16x16x32_bf16;
// W-panel staging via the Tensor Data Mover (tensor_load_to_lds + TENSORcnt).
typedef __attribute__((ext_vector_type(16))) __bf16 v16bf;
typedef __attribute__((ext_vector_type(8)))  __bf16 v8bf;
typedef __attribute__((ext_vector_type(8)))  float  v8f;
typedef __attribute__((ext_vector_type(4)))  unsigned int u32x4;
typedef __attribute__((ext_vector_type(8)))  int i32x8;
typedef __attribute__((ext_vector_type(4)))  int i32x4;

namespace ca {
constexpr int B = 64, L = 50, P = 49, T = 20, D = 512, H = 8, DH = 64;
constexpr int BL = B * L;                       // 3200
constexpr long ROWS_IMG = (long)BL * T;         // 64000  (out_img rows)
constexpr long ROWS_TIT = (long)BL * P;         // 156800 (out_title rows)
constexpr long ROWS     = ROWS_IMG + ROWS_TIT;  // 220800, divisible by 32

// ---- dynamic LDS layout for the attention kernel (per block) ----
constexpr int OFF_IMG_RM = 0;                        // [H][49][64] A-op of corr (pre-scaled by 1/8)
constexpr int OFF_TIT_RM = OFF_IMG_RM + H * 49 * 64; // [H][20][64] B-op of corr
constexpr int OFF_IMG_T  = OFF_TIT_RM + H * 20 * 64; // [H][64][64] B-op of hidden_img (d-major)
constexpr int OFF_TIT_T  = OFF_IMG_T  + H * 64 * 64; // [H][64][32] B-op of hidden_title (d-major)
constexpr int OFF_PIMG   = OFF_TIT_T  + H * 64 * 32; // [H][64][32] softmax(img) A-op
constexpr int OFF_PTIT   = OFF_PIMG   + H * 64 * 32; // [H][32][64] softmax(title) A-op
constexpr int N_BF16     = OFF_PTIT   + H * 32 * 64;
constexpr int OFF_CORR_B = (N_BF16 * 2 + 15) & ~15;           // [H][64][20] f32 corr scratch
constexpr int OFF_MASK_B = OFF_CORR_B + H * 64 * 20 * 4;      // [20] int mask
constexpr int SMEM_BYTES = (OFF_MASK_B + T * 4 + 127) & ~127; // ~275.5 KB (<= 320 KB/WGP)

// ---- GEMM kernel LDS: double-buffered 64x512 bf16 W panel (2 x 64 KB) ----
constexpr int PAN_BYTES  = 64 * D * 2;          // 65536
constexpr int GEMM_SMEM  = 2 * PAN_BYTES;       // 131072
} // namespace ca

// Load a 16x32 bf16 WMMA operand fragment (A-layout; B uses the same layout with
// "row" = N column of B, i.e. a row of the transposed-staged matrix).
// ISA 7.12.2: lanes 0-15 hold K 0..7 (VGPR0-3) / 16..23 (VGPR4-7) of row (lane&15);
// lanes 16-31 hold K 8..15 / 24..31.
__device__ __forceinline__ v16bf load_frag(const __bf16* base, long row, long stride,
                                           int k0, int lane) {
  const __bf16* p = base + row * stride + k0 + (((lane >> 4) & 1) << 3);
  union { v16bf v; v8bf h[2]; } u;
  u.h[0] = *(const v8bf*)(p);        // 16B -> VGPR0-3
  u.h[1] = *(const v8bf*)(p + 16);   // 16B -> VGPR4-7
  return u.v;
}

__device__ __forceinline__ v8f wmma_bf16(v16bf a, v16bf b, v8f c) {
  return __builtin_amdgcn_wmma_f32_16x16x32_bf16(false, a, false, b, (short)0, c,
                                                 false, false);
}

// TDM: DMA one 64-row W panel (64 KB contiguous) into LDS. D# per ISA ch.8:
// 1-D tile, data_size=4B, tensor_dim0 = tile_dim0 = 16384 elements, type=2.
// Groups 2/3 unused (zero); this toolchain's builtin takes 6 args (extra i32x8).
// Tracked with TENSORcnt.
__device__ __forceinline__ void tdm_load_panel(const __bf16* src, __bf16* dst_lds) {
  unsigned long long ga = (unsigned long long)(uintptr_t)src;     // 57-bit global
  unsigned lds = (unsigned)(uintptr_t)dst_lds;                    // low 32 = LDS offset
  u32x4 g0;
  g0[0] = 1u;                                                     // count=1, user desc
  g0[1] = lds;                                                    // lds_addr
  g0[2] = (unsigned)ga;                                           // global_addr[31:0]
  g0[3] = (unsigned)((ga >> 32) & 0x1FFFFFFu) | 0x80000000u;      // addr[56:32] | type=2
  i32x8 g1;
  g1[0] = 0x20000;                    // workgroup_mask=0, data_size=2 (4B), no flags
  g1[1] = (int)(16384u << 16);        // tensor_dim0[15:0]=16384 in bits[63:48]
  g1[2] = 0x10000;                    // tensor_dim0[31:16]=0, tensor_dim1=1
  g1[3] = (int)(16384u << 16);        // tile_dim0=16384 in bits[127:112]
  g1[4] = 0;                          // tile_dim1=0 (unused), tile_dim2=0
  g1[5] = 16384;                      // tensor_dim0_stride[31:0]
  g1[6] = 0;                          // stride hi, tensor_dim1_stride lo
  g1[7] = 0;
  i32x4 gz4 = {0, 0, 0, 0};
  i32x8 gz8 = {0, 0, 0, 0, 0, 0, 0, 0};
  __builtin_amdgcn_tensor_load_to_lds(g0, g1, gz4, gz4, gz8, 0);
}

// ---------------------------------------------------------------------------
// Kernel 1: W (f32, DxD) -> bf16 once. W stays L2-resident (512 KB) afterwards.
// ---------------------------------------------------------------------------
__global__ void ca_wconv(const float* __restrict__ W, __bf16* __restrict__ Wb, int n) {
  int i = blockIdx.x * blockDim.x + threadIdx.x;
  if (i < n) Wb[i] = (__bf16)W[i];
}

// ---------------------------------------------------------------------------
// Kernel 2: attention. One block per (b,l); wave w handles head w.
// Writes combined bf16 hidden matrix: rows [0, 64000) = hid_img (b,l,t,:),
// rows [64000, 220800) = hid_title (b,l,p,:), matching d_out's flat order.
// ---------------------------------------------------------------------------
__global__ void __launch_bounds__(256)
ca_attn(const float* __restrict__ img, const float* __restrict__ tit,
        const int* __restrict__ mask, const float* __restrict__ s_img,
        const float* __restrict__ s_tit, __bf16* __restrict__ hid) {
  using namespace ca;
  extern __shared__ char smem[];
  __bf16* bfb    = (__bf16*)smem;
  __bf16* img_rm = bfb + OFF_IMG_RM;
  __bf16* tit_rm = bfb + OFF_TIT_RM;
  __bf16* img_T  = bfb + OFF_IMG_T;
  __bf16* tit_T  = bfb + OFF_TIT_T;
  __bf16* pimg   = bfb + OFF_PIMG;
  __bf16* ptit   = bfb + OFF_PTIT;
  float*  corr_s = (float*)(smem + OFF_CORR_B);
  int*    mask_s = (int*)(smem + OFF_MASK_B);

  const int tid  = threadIdx.x;
  const int wave = tid >> 5;   // head
  const int lane = tid & 31;
  const int h    = wave;
  const int bl   = blockIdx.x;

  const float* img_g = img + (size_t)bl * P * D + h * DH;
  const float* tit_g = tit + (size_t)bl * T * D + h * DH;

  // ---- Phase 0: stage + convert; build row-major and d-major (transposed) copies.
  for (int p = 0; p < P; ++p) {
    float v0 = img_g[(size_t)p * D + lane];
    float v1 = img_g[(size_t)p * D + lane + 32];
    img_rm[(h * 49 + p) * 64 + lane]      = (__bf16)(v0 * 0.125f); // fold 1/sqrt(DH)
    img_rm[(h * 49 + p) * 64 + lane + 32] = (__bf16)(v1 * 0.125f);
    img_T[(h * 64 + lane) * 64 + p]        = (__bf16)v0;
    img_T[(h * 64 + lane + 32) * 64 + p]   = (__bf16)v1;
  }
  for (int p = P; p < 64; ++p) {  // zero K-padding for hidden_img
    img_T[(h * 64 + lane) * 64 + p]      = (__bf16)0.f;
    img_T[(h * 64 + lane + 32) * 64 + p] = (__bf16)0.f;
  }
  for (int t = 0; t < T; ++t) {
    float v0 = tit_g[(size_t)t * D + lane];
    float v1 = tit_g[(size_t)t * D + lane + 32];
    tit_rm[(h * 20 + t) * 64 + lane]      = (__bf16)v0;
    tit_rm[(h * 20 + t) * 64 + lane + 32] = (__bf16)v1;
    tit_T[(h * 64 + lane) * 32 + t]        = (__bf16)v0;
    tit_T[(h * 64 + lane + 32) * 32 + t]   = (__bf16)v1;
  }
  for (int t = T; t < 32; ++t) {  // zero K-padding for hidden_title
    tit_T[(h * 64 + lane) * 32 + t]      = (__bf16)0.f;
    tit_T[(h * 64 + lane + 32) * 32 + t] = (__bf16)0.f;
  }
  if (wave == 0 && lane < T) mask_s[lane] = mask[(size_t)bl * T + lane];
  __syncthreads();

  // ---- Phase 1: corr = (img/8) @ tit^T  (49x20, padded 64x32), K = 64.
  const int khalf = (lane >> 4) & 1;
  for (int mt = 0; mt < 4; ++mt) {
    for (int nt = 0; nt < 2; ++nt) {
      v8f c = {};
      int m = mt * 16 + (lane & 15); if (m > P - 1) m = P - 1;  // M-pad (discarded)
      int t = nt * 16 + (lane & 15); if (t > T - 1) t = T - 1;  // N-pad (discarded)
#pragma unroll
      for (int kc = 0; kc < 2; ++kc) {
        v16bf a = load_frag(img_rm + h * 49 * 64, m, 64, kc * 32, lane);
        v16bf b = load_frag(tit_rm + h * 20 * 64, t, 64, kc * 32, lane);
        c = wmma_bf16(a, b, c);
      }
      int colN = nt * 16 + (lane & 15);
      if (colN < T) {
        int rbase = mt * 16 + khalf * 8;   // C layout: M = rbase + r, N = colN
#pragma unroll
        for (int r = 0; r < 8; ++r) corr_s[(h * 64 + rbase + r) * 20 + colN] = c[r];
      }
    }
  }
  __syncthreads();

  // ---- Phase 2: softmaxes (VALU). Matches reference masking semantics.
  for (int p = lane; p < P; p += 32) {           // softmax over T per img row
    float sc = s_img[h * P + p];
    float mx = -3.4e38f;
    for (int t = 0; t < T; ++t) {
      float x = (mask_s[t] == 0) ? -1e9f : corr_s[(h * 64 + p) * 20 + t] * sc;
      mx = fmaxf(mx, x);
    }
    float sum = 0.f;
    for (int t = 0; t < T; ++t) {
      float x = (mask_s[t] == 0) ? -1e9f : corr_s[(h * 64 + p) * 20 + t] * sc;
      sum += __expf(x - mx);
    }
    float inv = 1.f / sum;
    for (int t = 0; t < T; ++t) {
      float x = (mask_s[t] == 0) ? -1e9f : corr_s[(h * 64 + p) * 20 + t] * sc;
      pimg[(h * 64 + p) * 32 + t] = (__bf16)(__expf(x - mx) * inv);
    }
    for (int t = T; t < 32; ++t) pimg[(h * 64 + p) * 32 + t] = (__bf16)0.f; // K-pad
  }
  if (lane < T) {                                 // softmax over P per title row
    int t = lane;
    float sc = s_tit[h * T + t];
    bool masked = (mask_s[t] == 0);               // whole row -> uniform 1/49
    float mx = -3.4e38f;
    for (int p = 0; p < P; ++p) {
      float x = masked ? -1e9f : corr_s[(h * 64 + p) * 20 + t] * sc;
      mx = fmaxf(mx, x);
    }
    float sum = 0.f;
    for (int p = 0; p < P; ++p) {
      float x = masked ? -1e9f : corr_s[(h * 64 + p) * 20 + t] * sc;
      sum += __expf(x - mx);
    }
    float inv = 1.f / sum;
    for (int p = 0; p < P; ++p) {
      float x = masked ? -1e9f : corr_s[(h * 64 + p) * 20 + t] * sc;
      ptit[(h * 32 + t) * 64 + p] = (__bf16)(__expf(x - mx) * inv);
    }
    for (int p = P; p < 64; ++p) ptit[(h * 32 + t) * 64 + p] = (__bf16)0.f;  // K-pad
  }
  __syncthreads();

  // ---- Phase 3: hidden_title = p_img(49x20) @ tit(20x64), K padded to 32.
  const long row_tit = ROWS_IMG + (long)bl * P;   // rows in combined hidden matrix
  for (int mt = 0; mt < 4; ++mt) {
    for (int nt = 0; nt < 4; ++nt) {
      v8f c = {};
      v16bf a = load_frag(pimg + h * 64 * 32, mt * 16 + (lane & 15), 32, 0, lane);
      v16bf b = load_frag(tit_T + h * 64 * 32, nt * 16 + (lane & 15), 32, 0, lane);
      c = wmma_bf16(a, b, c);
      int colN = h * DH + nt * 16 + (lane & 15);
      int rbase = mt * 16 + khalf * 8;
#pragma unroll
      for (int r = 0; r < 8; ++r) {
        int p = rbase + r;
        if (p < P) hid[(row_tit + p) * D + colN] = (__bf16)c[r];
      }
    }
  }

  // ---- Phase 4: hidden_img = p_title(20x49) @ img(49x64), K padded to 64.
  const long row_img = (long)bl * T;
  for (int mt = 0; mt < 2; ++mt) {
    for (int nt = 0; nt < 4; ++nt) {
      v8f c = {};
#pragma unroll
      for (int kc = 0; kc < 2; ++kc) {
        v16bf a = load_frag(ptit + h * 32 * 64, mt * 16 + (lane & 15), 64, kc * 32, lane);
        v16bf b = load_frag(img_T + h * 64 * 64, nt * 16 + (lane & 15), 64, kc * 32, lane);
        c = wmma_bf16(a, b, c);
      }
      int colN = h * DH + nt * 16 + (lane & 15);
      int rbase = mt * 16 + khalf * 8;
#pragma unroll
      for (int r = 0; r < 8; ++r) {
        int t = rbase + r;
        if (t < T) hid[(row_img + t) * D + colN] = (__bf16)c[r];
      }
    }
  }
}

// ---------------------------------------------------------------------------
// Kernel 3: out = hid(220800x512, bf16) @ W^T + bias, f32 out straight to d_out.
// Block = 8 waves; each wave owns a 32(M) x 64(N) tile. W panels (64x512 bf16)
// are TDM-staged into LDS, double-buffered: issue panel nc+1, compute panel nc
// from LDS, s_wait_tensorcnt, barrier. All waves stay for barriers (uniform).
// ---------------------------------------------------------------------------
__global__ void __launch_bounds__(256)
ca_gemm(const __bf16* __restrict__ A, const __bf16* __restrict__ Wb,
        const float* __restrict__ bias, float* __restrict__ out) {
  using namespace ca;
  extern __shared__ char smem[];
  __bf16* pan0 = (__bf16*)smem;                       // 64 x 512 bf16
  __bf16* pan1 = (__bf16*)(smem + PAN_BYTES);
  const int lane = threadIdx.x & 31;
  const int wave = threadIdx.x >> 5;
  const long m0 = ((long)blockIdx.x * 8 + wave) * 32; // ROWS % 32 == 0
  const bool active = (m0 < ROWS);
  const int khalf8 = ((lane >> 4) & 1) * 8;

  if (active) __builtin_prefetch(A + m0 * D, 0, 1);   // global_prefetch_b8

  if (wave == 0) {                                    // prologue: stage panel 0
    tdm_load_panel(Wb, pan0);
    __builtin_amdgcn_s_wait_tensorcnt(0);
  }
  __syncthreads();

  for (int nc = 0; nc < 8; ++nc) {
    __bf16* pan = (nc & 1) ? pan1 : pan0;
    __bf16* nxt = (nc & 1) ? pan0 : pan1;
    if (wave == 0 && nc < 7)                          // overlap DMA with compute
      tdm_load_panel(Wb + (size_t)(nc + 1) * 64 * D, nxt);

    if (active) {
      v8f acc[2][4] = {};
      for (int kc = 0; kc < 16; ++kc) {
        v16bf a0 = load_frag(A, m0 + (lane & 15), D, kc * 32, lane);
        v16bf a1 = load_frag(A, m0 + 16 + (lane & 15), D, kc * 32, lane);
#pragma unroll
        for (int nt = 0; nt < 4; ++nt) {
          // B[k][n] = W[n][k]: lane n reads LDS panel row (nt*16+n), K-contig.
          v16bf b = load_frag(pan, nt * 16 + (lane & 15), D, kc * 32, lane);
          acc[0][nt] = wmma_bf16(a0, b, acc[0][nt]);
          acc[1][nt] = wmma_bf16(a1, b, acc[1][nt]);
        }
      }
#pragma unroll
      for (int mi = 0; mi < 2; ++mi) {
#pragma unroll
        for (int nt = 0; nt < 4; ++nt) {
          int coln = nc * 64 + nt * 16 + (lane & 15);
          float bv = bias[coln];
          long rbase = m0 + mi * 16 + khalf8;
#pragma unroll
          for (int r = 0; r < 8; ++r)
            out[(rbase + r) * D + coln] = acc[mi][nt][r] + bv;
        }
      }
    }

    if (wave == 0) __builtin_amdgcn_s_wait_tensorcnt(0); // next panel landed
    __syncthreads();
  }
}

// ---------------------------------------------------------------------------
extern "C" void kernel_launch(void* const* d_in, const int* in_sizes, int n_in,
                              void* d_out, int out_size, void* d_ws, size_t ws_size,
                              hipStream_t stream) {
  using namespace ca;
  const float* img   = (const float*)d_in[0];
  const float* tit   = (const float*)d_in[1];
  const int*   mask  = (const int*)d_in[2];
  const float* s_img = (const float*)d_in[3];
  const float* s_tit = (const float*)d_in[4];
  const float* W     = (const float*)d_in[5];
  const float* bias  = (const float*)d_in[6];
  float* out = (float*)d_out;

  // Workspace: [W bf16: D*D][hidden bf16: ROWS*D] ~= 226.6 MB total.
  __bf16* Wb  = (__bf16*)d_ws;
  __bf16* hid = Wb + (size_t)D * D;

  ca_wconv<<<dim3((D * D + 255) / 256), dim3(256), 0, stream>>>(W, Wb, D * D);
  ca_attn<<<dim3(BL), dim3(256), SMEM_BYTES, stream>>>(img, tit, mask, s_img, s_tit, hid);
  const int gemm_blocks = (int)((ROWS / 32 + 7) / 8);  // 863
  ca_gemm<<<dim3(gemm_blocks), dim3(256), GEMM_SMEM, stream>>>(hid, Wb, bias, out);
}
